// PharmMatchNetFast_90237262889462
// MI455X (gfx1250) — matrix operand order repaired and checked by reading, not compiled, not run on hardware
//
#include <hip/hip_runtime.h>
#include <math.h>

// ---------------------------------------------------------------------------
// PharmMatchNet fused pipeline for MI455X (gfx1250, wave32, WMMA bf16)
// ---------------------------------------------------------------------------

typedef __attribute__((ext_vector_type(16))) __bf16 v16bf;
typedef __attribute__((ext_vector_type(8)))  __bf16 v8bf;
typedef __attribute__((ext_vector_type(8)))  float  v8f;

#define DMODEL 256
#define AP 264                 // padded LDS row stride (bf16/f32 elems) for 16xK tiles
#define N_L 4096
#define N_P 8192
#define PX_K 192
#define RETR 128
#define SCALE_LOGITS 0.0625f   // 1/sqrt(256)

__device__ __forceinline__ float gelu_f(float x) {
  return 0.5f * x * (1.0f + erff(x * 0.70710678118654752440f));
}

__device__ __forceinline__ v8f wmma_bf16(v16bf a, v16bf b, v8f c) {
  // (neg_a, A, neg_b, B, c_mod, C, reuse_a, reuse_b)
  return __builtin_amdgcn_wmma_f32_16x16x32_bf16(false, a, false, b, (short)0, c,
                                                 false, false);
}

// A fragment (16x32 bf16) from a row-major [16 x K] buffer, row stride lda.
// Layout used here (matches B): lane l holds row (l&15); K = 16*(l>>4) + {0..15} in elems 0..15
// (same order as B)  -> two contiguous 16B loads.
__device__ __forceinline__ v16bf load_a_rm(const __bf16* A, int lda, int ks) {
  int lane = threadIdx.x & 31;
  int m = lane & 15;
  int kb = ((lane >> 4) << 4) + (ks << 5);
  const __bf16* p = A + m * lda + kb;
  v8bf lo = *(const v8bf*)(p);
  v8bf hi = *(const v8bf*)(p + 8);
  return __builtin_shufflevector(lo, hi, 0,1,2,3,4,5,6,7,8,9,10,11,12,13,14,15);
}

// B fragment (32x16 bf16) with B[k][n] = BT[n][k]; BT row-major [N x K], stride ldk.
// Layout used here (matches A): lane l holds column n0+(l&15); K = 16*(l>>4) + 0..15 -> one 32B load.
__device__ __forceinline__ v16bf load_b_t(const __bf16* BT, int ldk, int n0, int ks) {
  int lane = threadIdx.x & 31;
  int n = n0 + (lane & 15);
  int kb = ((lane >> 4) << 4) + (ks << 5);
  return *(const v16bf*)(BT + n * ldk + kb);
}

// One wave computes two adjacent 16x16 output tiles (col tiles ct0, ct0+1).
__device__ __forceinline__ void gemm_2ct(const __bf16* A, int lda, const __bf16* BT,
                                         int ldk, int ksteps, int ct0,
                                         v8f& acc0, v8f& acc1) {
  for (int ks = 0; ks < ksteps; ++ks) {
    v16bf a  = load_a_rm(A, lda, ks);
    v16bf b0 = load_b_t(BT, ldk, ct0 * 16, ks);
    acc0 = wmma_bf16(a, b0, acc0);
    v16bf b1 = load_b_t(BT, ldk, ct0 * 16 + 16, ks);
    acc1 = wmma_bf16(a, b1, acc1);
  }
}

// C/D layout: VGPR v holds M = v + 8*(lane>=16), N = lane&15.
__device__ __forceinline__ void acc_store(float* H, int ldh, v8f acc, int ct) {
  int lane = threadIdx.x & 31;
  int col = ct * 16 + (lane & 15);
  int rb = (lane >> 4) << 3;
  #pragma unroll
  for (int v = 0; v < 8; ++v) H[(rb + v) * ldh + col] = acc[v];
}

// LayerNorm 16x256 tile: H (f32) -> A (bf16). 16 threads cooperate per row.
__device__ __forceinline__ void ln_to_bf16(float (*H)[AP], __bf16 (*A)[AP],
                                           const float* g, const float* be,
                                           float* redM, float* redR) {
  int t = threadIdx.x;
  int r = t >> 4, j = t & 15;
  float s = 0.f, s2 = 0.f;
  for (int c = j; c < DMODEL; c += 16) { float x = H[r][c]; s += x; s2 += x * x; }
  #pragma unroll
  for (int off = 8; off > 0; off >>= 1) {
    s += __shfl_xor(s, off, 16);
    s2 += __shfl_xor(s2, off, 16);
  }
  if (j == 0) {
    float mu = s * (1.0f / DMODEL);
    redM[r] = mu;
    redR[r] = rsqrtf(s2 * (1.0f / DMODEL) - mu * mu + 1e-5f);
  }
  __syncthreads();
  int col = t;
  float gc = g[col], bc = be[col];
  for (int rr = 0; rr < 16; ++rr)
    A[rr][col] = (__bf16)((H[rr][col] - redM[rr]) * redR[rr] * gc + bc);
}

__device__ __forceinline__ float blk_sum(float v, float* sh) {
  int t = threadIdx.x;
  sh[t] = v; __syncthreads();
  for (int s = 128; s > 0; s >>= 1) { if (t < s) sh[t] += sh[t + s]; __syncthreads(); }
  float r = sh[0]; __syncthreads();
  return r;
}

// --------------------------- weight prep -----------------------------------
// f32 [K][N] -> bf16 [N][K] (transposed so B-fragments are contiguous 32B loads)
__global__ void transpose_to_bf16_kernel(const float* __restrict__ src,
                                         __bf16* __restrict__ dst, int K, int N) {
  int idx = blockIdx.x * 256 + threadIdx.x;
  if (idx >= K * N) return;
  int n = idx / K, k = idx - n * K;
  dst[idx] = (__bf16)src[k * N + n];
}

// --------------------------- landmark embed --------------------------------
__global__ __launch_bounds__(256) void embed_landmarks_kernel(
    const float* __restrict__ l_x, const int* __restrict__ l_typ,
    const float* __restrict__ le_emb, const float* __restrict__ le_w1,
    const float* __restrict__ le_b1, const float* __restrict__ le_b2,
    const float* __restrict__ le_g, const float* __restrict__ le_be,
    const __bf16* __restrict__ le_w2T, const __bf16* __restrict__ proj_lT,
    float* __restrict__ hL, __bf16* __restrict__ hLb) {
  __shared__ __bf16 Atile[16][AP];
  __shared__ float  Htile[16][AP];
  __shared__ float  redM[16], redR[16];
  int t = threadIdx.x, wave = t >> 5;
  int row0 = blockIdx.x * 16;

  { // layer-1 (K=8) scalar: one output column per thread
    float w1c[8], b = le_b1[t];
    #pragma unroll
    for (int k = 0; k < 8; ++k) w1c[k] = le_w1[k * DMODEL + t];
    for (int r = 0; r < 16; ++r) {
      float a = b;
      #pragma unroll
      for (int k = 0; k < 8; ++k) a += l_x[(row0 + r) * 8 + k] * w1c[k];
      Atile[r][t] = (__bf16)gelu_f(a);
    }
  }
  __syncthreads();
  { // t1 @ le_w2
    v8f a0 = {}, a1 = {};
    gemm_2ct(&Atile[0][0], AP, le_w2T, DMODEL, 8, wave * 2, a0, a1);
    acc_store(&Htile[0][0], AP, a0, wave * 2);
    acc_store(&Htile[0][0], AP, a1, wave * 2 + 1);
  }
  __syncthreads();
  { // + b2 + type embedding
    float b2 = le_b2[t];
    for (int r = 0; r < 16; ++r) {
      int ty = l_typ[row0 + r]; ty = ty < 0 ? 0 : (ty > 255 ? 255 : ty);
      Htile[r][t] += b2 + le_emb[ty * DMODEL + t];
    }
  }
  __syncthreads();
  ln_to_bf16(Htile, Atile, le_g, le_be, redM, redR);
  __syncthreads();
  { // @ proj_l
    v8f a0 = {}, a1 = {};
    gemm_2ct(&Atile[0][0], AP, proj_lT, DMODEL, 8, wave * 2, a0, a1);
    acc_store(&Htile[0][0], AP, a0, wave * 2);
    acc_store(&Htile[0][0], AP, a1, wave * 2 + 1);
  }
  __syncthreads();
  for (int r = 0; r < 16; ++r) { // coalesced global writeout (f32 + bf16 mirror)
    float v = Htile[r][t];
    hL [(size_t)(row0 + r) * DMODEL + t] = v;
    hLb[(size_t)(row0 + r) * DMODEL + t] = (__bf16)v;
  }
}

// --------------------------- point embed -----------------------------------
__global__ __launch_bounds__(256) void embed_points_kernel(
    const float* __restrict__ p_x, const float* __restrict__ p_score,
    const float* __restrict__ p_rad, const int* __restrict__ p_typ,
    const float* __restrict__ pe_emb, const float* __restrict__ pe_b1,
    const float* __restrict__ pe_b2, const float* __restrict__ ps_w1,
    const float* __restrict__ ps_b1, const float* __restrict__ ps_b2,
    const float* __restrict__ pe_g, const float* __restrict__ pe_be,
    const float* __restrict__ sig_b1, const float* __restrict__ sig_w2,
    const float* __restrict__ sig_b2,
    const __bf16* __restrict__ pe_w1T, const __bf16* __restrict__ pe_w2T,
    const __bf16* __restrict__ ps_w2T, const __bf16* __restrict__ proj_pT,
    const __bf16* __restrict__ sig_w1T,
    float* __restrict__ hP, __bf16* __restrict__ hPb, float* __restrict__ p_sigma) {
  __shared__ __bf16 Atile[16][AP];
  __shared__ __bf16 Stile[16][AP];
  __shared__ float  Htile[16][AP];
  __shared__ float  redM[16], redR[16];
  int t = threadIdx.x, wave = t >> 5;
  int row0 = blockIdx.x * 16;

  for (int idx = t; idx < 16 * PX_K; idx += 256) { // stage p_x tile as bf16
    int r = idx / PX_K, c = idx - r * PX_K;
    Atile[r][c] = (__bf16)p_x[(size_t)(row0 + r) * PX_K + c];
  }
  { // score branch hidden: g1 = gelu(score * ps_w1 + ps_b1)
    float w = ps_w1[t], b = ps_b1[t];
    for (int r = 0; r < 16; ++r)
      Stile[r][t] = (__bf16)gelu_f(p_score[row0 + r] * w + b);
  }
  __syncthreads();
  { // u = p_x @ pe_w1   (K = 192 -> 6 k-steps)
    v8f a0 = {}, a1 = {};
    gemm_2ct(&Atile[0][0], AP, pe_w1T, PX_K, 6, wave * 2, a0, a1);
    acc_store(&Htile[0][0], AP, a0, wave * 2);
    acc_store(&Htile[0][0], AP, a1, wave * 2 + 1);
  }
  __syncthreads();
  { // t1 = gelu(u + pe_b1) -> Atile
    float b = pe_b1[t];
    for (int r = 0; r < 16; ++r) Atile[r][t] = (__bf16)gelu_f(Htile[r][t] + b);
  }
  __syncthreads();
  { // h = t1@pe_w2 + g1@ps_w2  (two chains into the same accumulators)
    v8f a0 = {}, a1 = {};
    gemm_2ct(&Atile[0][0], AP, pe_w2T, DMODEL, 8, wave * 2, a0, a1);
    gemm_2ct(&Stile[0][0], AP, ps_w2T, DMODEL, 8, wave * 2, a0, a1);
    acc_store(&Htile[0][0], AP, a0, wave * 2);
    acc_store(&Htile[0][0], AP, a1, wave * 2 + 1);
  }
  __syncthreads();
  {
    float b = pe_b2[t] + ps_b2[t];
    for (int r = 0; r < 16; ++r) {
      int ty = p_typ[row0 + r]; ty = ty < 0 ? 0 : (ty > 255 ? 255 : ty);
      Htile[r][t] += b + pe_emb[ty * DMODEL + t];
    }
  }
  __syncthreads();
  ln_to_bf16(Htile, Atile, pe_g, pe_be, redM, redR);
  __syncthreads();
  { // hP = LN @ proj_p
    v8f a0 = {}, a1 = {};
    gemm_2ct(&Atile[0][0], AP, proj_pT, DMODEL, 8, wave * 2, a0, a1);
    acc_store(&Htile[0][0], AP, a0, wave * 2);
    acc_store(&Htile[0][0], AP, a1, wave * 2 + 1);
  }
  __syncthreads();
  for (int r = 0; r < 16; ++r) {
    float v = Htile[r][t];
    hP [(size_t)(row0 + r) * DMODEL + t] = v;
    hPb[(size_t)(row0 + r) * DMODEL + t] = (__bf16)v;
    Atile[r][t] = (__bf16)v;                 // reuse hP tile for sigma head
  }
  __syncthreads();
  { // sigma hidden = hP @ sig_w1
    v8f a0 = {}, a1 = {};
    gemm_2ct(&Atile[0][0], AP, sig_w1T, DMODEL, 8, wave * 2, a0, a1);
    acc_store(&Htile[0][0], AP, a0, wave * 2);
    acc_store(&Htile[0][0], AP, a1, wave * 2 + 1);
  }
  __syncthreads();
  { // log_sig = gelu(hidden + b1) @ sig_w2 + b2; p_sigma = softplus + eps + rad
    int r = t >> 4, j = t & 15;
    float s = 0.f;
    for (int c = j; c < DMODEL; c += 16) s += gelu_f(Htile[r][c] + sig_b1[c]) * sig_w2[c];
    #pragma unroll
    for (int off = 8; off > 0; off >>= 1) s += __shfl_xor(s, off, 16);
    if (j == 0) {
      float ls = s + sig_b2[0];
      float sp = ls > 20.f ? ls : log1pf(expf(ls));
      float pr = p_rad[row0 + r];
      p_sigma[row0 + r] = sp + 1e-3f + (pr > 0.f ? pr : 0.f);
    }
  }
}

// --------------------------- logits GEMM -----------------------------------
// Block = 8 waves computes a 32(M) x 128(N) tile; B fragments shared across
// the two M-subtiles. 128MB output stays resident in the 192MB L2.
__global__ __launch_bounds__(256) void logits_kernel(
    const __bf16* __restrict__ hLb, const __bf16* __restrict__ hPb,
    float* __restrict__ logits) {
  int lane = threadIdx.x & 31, wave = threadIdx.x >> 5;
  int m0 = blockIdx.y * 32;
  int n0 = blockIdx.x * 128 + wave * 16;
  const __bf16* A0 = hLb + (size_t)m0 * DMODEL;
  const __bf16* A1 = A0 + 16 * DMODEL;
  v8f acc0 = {}, acc1 = {};
  #pragma unroll
  for (int ks = 0; ks < 8; ++ks) {
    v16bf b  = load_b_t(hPb, DMODEL, n0, ks);   // hP rows ARE the B columns
    v16bf a0 = load_a_rm(A0, DMODEL, ks);
    acc0 = wmma_bf16(a0, b, acc0);
    v16bf a1 = load_a_rm(A1, DMODEL, ks);
    acc1 = wmma_bf16(a1, b, acc1);
  }
  int col = n0 + (lane & 15);
  int rb = (lane >> 4) << 3;
  #pragma unroll
  for (int v = 0; v < 8; ++v) {
    logits[(size_t)(m0 + rb + v) * N_P + col]      = acc0[v] * SCALE_LOGITS;
    logits[(size_t)(m0 + 16 + rb + v) * N_P + col] = acc1[v] * SCALE_LOGITS;
  }
}

// --------------------------- per-row stats + top-8 -------------------------
__global__ __launch_bounds__(256) void row_stats_kernel(
    const float* __restrict__ logits, const float* __restrict__ hL,
    const float* __restrict__ dustbin,
    float* __restrict__ sharp_rows, float* __restrict__ ent_rows,
    int* __restrict__ topj, float* __restrict__ logWtop) {
  __shared__ float sval[256];
  __shared__ int   sidx[256];
  __shared__ float tv[8];
  __shared__ int   tix[8];
  int r = blockIdx.x, t = threadIdx.x;

  // dustbin logit for this row
  sval[t] = hL[(size_t)r * DMODEL + t] * dustbin[t];
  __syncthreads();
  for (int s = 128; s > 0; s >>= 1) { if (t < s) sval[t] += sval[t + s]; __syncthreads(); }
  float dust = sval[0] * SCALE_LOGITS;
  __syncthreads();

  float v[32];
  const float* row = logits + (size_t)r * N_P;
  #pragma unroll
  for (int i = 0; i < 32; ++i) v[i] = row[t + (i << 8)];

  float m = dust;
  #pragma unroll
  for (int i = 0; i < 32; ++i) m = fmaxf(m, v[i]);
  sval[t] = m; __syncthreads();
  for (int s = 128; s > 0; s >>= 1) { if (t < s) sval[t] = fmaxf(sval[t], sval[t + s]); __syncthreads(); }
  m = sval[0]; __syncthreads();

  float s0 = 0.f, s1 = 0.f;
  #pragma unroll
  for (int i = 0; i < 32; ++i) { float d = v[i] - m; float e = __expf(d); s0 += e; s1 += d * e; }
  sval[t] = s0; __syncthreads();
  for (int s = 128; s > 0; s >>= 1) { if (t < s) sval[t] += sval[t + s]; __syncthreads(); }
  float S0 = sval[0]; __syncthreads();
  sval[t] = s1; __syncthreads();
  for (int s = 128; s > 0; s >>= 1) { if (t < s) sval[t] += sval[t + s]; __syncthreads(); }
  float S1 = sval[0]; __syncthreads();
  { float d = dust - m, e = __expf(d); S0 += e; S1 += d * e; }
  float logS0 = logf(S0);
  if (t == 0) {
    sharp_rows[r] = 1.0f / S0;           // max Wm = exp(m - m)/S0
    ent_rows[r]   = logS0 - S1 / S0;     // -(sum W logW)
  }

  // exact top-8 via 8 argmax rounds (row stays hot in registers)
  for (int round = 0; round < 8; ++round) {
    float best = -3.0e38f; int bi = 0;
    #pragma unroll
    for (int i = 0; i < 32; ++i) if (v[i] > best) { best = v[i]; bi = t + (i << 8); }
    sval[t] = best; sidx[t] = bi; __syncthreads();
    for (int s = 128; s > 0; s >>= 1) {
      if (t < s && sval[t + s] > sval[t]) { sval[t] = sval[t + s]; sidx[t] = sidx[t + s]; }
      __syncthreads();
    }
    if (t == 0) { tv[round] = sval[0]; tix[round] = sidx[0]; }
    __syncthreads();
    int win = tix[round];
    if ((win & 255) == t) v[win >> 8] = -3.0e38f;
    __syncthreads();
  }
  if (t < 8) {
    topj[r * 8 + t] = tix[t];
    logWtop[r * 8 + t] = tv[t] - m - logS0;
  }
}

// --------------------------- column sums (global means) --------------------
__global__ __launch_bounds__(256) void colsum_kernel(
    const float* __restrict__ hL, const float* __restrict__ hP,
    float* __restrict__ lg, float* __restrict__ pg) {
  int b = blockIdx.x, t = threadIdx.x;
  if (b < N_L / 16) {
    float s = 0.f; int r0 = b * 16;
    for (int r = 0; r < 16; ++r) s += hL[(size_t)(r0 + r) * DMODEL + t];
    atomicAdd(&lg[t], s);
  } else {
    float s = 0.f; int r0 = (b - N_L / 16) * 16;
    for (int r = 0; r < 16; ++r) s += hP[(size_t)(r0 + r) * DMODEL + t];
    atomicAdd(&pg[t], s);
  }
}

// --------------------------- finalize --------------------------------------
__global__ __launch_bounds__(256) void finalize_kernel(
    const float* __restrict__ sharp_r, const float* __restrict__ ent_r,
    const float* __restrict__ lg, const float* __restrict__ pg,
    const float* __restrict__ retr_l, const float* __restrict__ retr_p,
    const float* __restrict__ l_pos, const float* __restrict__ p_pos,
    const int* __restrict__ pairs, const int* __restrict__ topj,
    const float* __restrict__ logWtop, const float* __restrict__ p_sigma,
    const float* __restrict__ hi_w1, const float* __restrict__ hi_b1,
    const float* __restrict__ hi_w2, const float* __restrict__ hi_b2,
    float* __restrict__ out) {
  __shared__ float sh[256];
  __shared__ float Lg[256], Pg[256];
  __shared__ float lr[128], pr[128];
  int t = threadIdx.x;

  float s = 0.f, e = 0.f;
  for (int r = t; r < N_L; r += 256) { s += sharp_r[r]; e += ent_r[r]; }
  float sharp = blk_sum(s, sh) * (1.0f / N_L);
  float ent   = blk_sum(e, sh) * (1.0f / N_L);

  Lg[t] = lg[t] * (1.0f / N_L);
  Pg[t] = pg[t] * (1.0f / N_P);
  __syncthreads();
  if (t < 128) {
    float a = 0.f;
    for (int k = 0; k < DMODEL; ++k) a += Lg[k] * retr_l[k * RETR + t];
    lr[t] = a;
  } else {
    int j = t - 128; float a = 0.f;
    for (int k = 0; k < DMODEL; ++k) a += Pg[k] * retr_p[k * RETR + j];
    pr[j] = a;
  }
  __syncthreads();
  float nl2 = blk_sum(t < 128 ? lr[t] * lr[t] : 0.f, sh);
  float np2 = blk_sum(t < 128 ? pr[t] * pr[t] : 0.f, sh);
  float dlp = blk_sum(t < 128 ? lr[t] * pr[t] : 0.f, sh);
  float retr_sim = dlp / (fmaxf(sqrtf(nl2), 1e-12f) * fmaxf(sqrtf(np2), 1e-12f));

  float lmix = 0.f;
  if (t < 64) {            // one edge pair per thread: 8x8 Gaussian mixture
    int ai = pairs[2 * t], bi = pairs[2 * t + 1];
    float dx = l_pos[ai * 3 + 0] - l_pos[bi * 3 + 0];
    float dy = l_pos[ai * 3 + 1] - l_pos[bi * 3 + 1];
    float dz = l_pos[ai * 3 + 2] - l_pos[bi * 3 + 2];
    float dab = sqrtf(fmaxf(dx * dx + dy * dy + dz * dz, 1e-12f));
    float mx = -3.0e38f, acc = 0.f;
    for (int j = 0; j < 8; ++j) {
      int pa = topj[ai * 8 + j];
      float ax = p_pos[pa * 3 + 0], ay = p_pos[pa * 3 + 1], az = p_pos[pa * 3 + 2];
      float sa = p_sigma[pa];
      float lwa = logWtop[ai * 8 + j];
      for (int k = 0; k < 8; ++k) {
        int pb = topj[bi * 8 + k];
        float bx = ax - p_pos[pb * 3 + 0];
        float by = ay - p_pos[pb * 3 + 1];
        float bz = az - p_pos[pb * 3 + 2];
        float mu = sqrtf(fmaxf(bx * bx + by * by + bz * bz, 1e-12f));
        float sb = p_sigma[pb];
        float sg = fmaxf(sqrtf(sa * sa + sb * sb + 1e-6f), 1e-4f);
        float z = (dab - mu) / sg;
        float x = lwa + logWtop[bi * 8 + k]
                - 0.5f * z * z - logf(sg) - 0.91893853320467274178f;
        if (x > mx) { acc = acc * __expf(mx - x) + 1.0f; mx = x; }
        else          acc += __expf(x - mx);
      }
    }
    lmix = mx + logf(acc);
  }
  float edge_raw = blk_sum(t < 64 ? lmix : 0.f, sh) * (1.0f / 64.0f);
  float edge = tanhf(edge_raw * (1.0f / 5.0f));

  float hp = 0.f;
  if (t < 64) {
    float h = sharp * hi_w1[t] + (-ent) * hi_w1[64 + t] + edge * hi_w1[128 + t] + hi_b1[t];
    hp = gelu_f(h) * hi_w2[t];
  }
  float score = blk_sum(hp, sh) + hi_b2[0];

  if (t == 0) {
    out[0] = score;
    out[1] = sharp;
    out[2] = -ent;
    out[3] = edge;
    out[4] = retr_sim;
  }
}

// --------------------------- host launcher ---------------------------------
extern "C" void kernel_launch(void* const* d_in, const int* in_sizes, int n_in,
                              void* d_out, int out_size, void* d_ws, size_t ws_size,
                              hipStream_t stream) {
  (void)in_sizes; (void)n_in; (void)out_size; (void)ws_size;
  const float* l_pos   = (const float*)d_in[0];
  const float* l_x     = (const float*)d_in[1];
  const int*   l_typ   = (const int*)  d_in[2];
  /* d_in[3]: l_mask (all-true in this workload) */
  const float* p_pos   = (const float*)d_in[4];
  const float* p_x     = (const float*)d_in[5];
  const int*   p_typ   = (const int*)  d_in[6];
  const float* p_score = (const float*)d_in[7];
  const float* p_rad   = (const float*)d_in[8];
  /* d_in[9]: p_mask (all-true) */
  const int*   pairs   = (const int*)  d_in[10];
  const float* le_emb = (const float*)d_in[11];
  const float* le_w1  = (const float*)d_in[12];
  const float* le_b1  = (const float*)d_in[13];
  const float* le_w2  = (const float*)d_in[14];
  const float* le_b2  = (const float*)d_in[15];
  const float* le_g   = (const float*)d_in[16];
  const float* le_be  = (const float*)d_in[17];
  const float* pe_emb = (const float*)d_in[18];
  const float* pe_w1  = (const float*)d_in[19];
  const float* pe_b1  = (const float*)d_in[20];
  const float* pe_w2  = (const float*)d_in[21];
  const float* pe_b2  = (const float*)d_in[22];
  const float* ps_w1  = (const float*)d_in[23];
  const float* ps_b1  = (const float*)d_in[24];
  const float* ps_w2  = (const float*)d_in[25];
  const float* ps_b2  = (const float*)d_in[26];
  const float* pe_g   = (const float*)d_in[27];
  const float* pe_be  = (const float*)d_in[28];
  const float* proj_l = (const float*)d_in[29];
  const float* proj_p = (const float*)d_in[30];
  const float* dustbin= (const float*)d_in[31];
  const float* sig_w1 = (const float*)d_in[32];
  const float* sig_b1 = (const float*)d_in[33];
  const float* sig_w2 = (const float*)d_in[34];
  const float* sig_b2 = (const float*)d_in[35];
  const float* hi_w1  = (const float*)d_in[36];
  const float* hi_b1  = (const float*)d_in[37];
  const float* hi_w2  = (const float*)d_in[38];
  const float* hi_b2  = (const float*)d_in[39];
  const float* retr_l = (const float*)d_in[40];
  const float* retr_p = (const float*)d_in[41];

  char* ws = (char*)d_ws;
  size_t off = 0;
  auto carve = [&](size_t bytes) -> void* {
    void* p = ws + off;
    off += (bytes + 255) & ~(size_t)255;
    return p;
  };
  float*  hL      = (float*) carve((size_t)N_L * DMODEL * 4);
  float*  hP      = (float*) carve((size_t)N_P * DMODEL * 4);
  __bf16* hLb     = (__bf16*)carve((size_t)N_L * DMODEL * 2);
  __bf16* hPb     = (__bf16*)carve((size_t)N_P * DMODEL * 2);
  float*  logits  = (float*) carve((size_t)N_L * N_P * 4);   // 128MB: L2-resident
  float*  p_sigma = (float*) carve((size_t)N_P * 4);
  float*  sharp_r = (float*) carve((size_t)N_L * 4);
  float*  ent_r   = (float*) carve((size_t)N_L * 4);
  int*    topj    = (int*)   carve((size_t)N_L * 8 * 4);
  float*  logWtop = (float*) carve((size_t)N_L * 8 * 4);
  float*  lg      = (float*) carve(512 * 4);
  float*  pg      = lg + 256;
  __bf16* le_w2T  = (__bf16*)carve(256 * 256 * 2);
  __bf16* proj_lT = (__bf16*)carve(256 * 256 * 2);
  __bf16* pe_w1T  = (__bf16*)carve(256 * 192 * 2);
  __bf16* pe_w2T  = (__bf16*)carve(256 * 256 * 2);
  __bf16* ps_w2T  = (__bf16*)carve(256 * 256 * 2);
  __bf16* proj_pT = (__bf16*)carve(256 * 256 * 2);
  __bf16* sig_w1T = (__bf16*)carve(256 * 256 * 2);

  transpose_to_bf16_kernel<<<256, 256, 0, stream>>>(le_w2,  le_w2T,  256, 256);
  transpose_to_bf16_kernel<<<256, 256, 0, stream>>>(proj_l, proj_lT, 256, 256);
  transpose_to_bf16_kernel<<<192, 256, 0, stream>>>(pe_w1,  pe_w1T,  192, 256);
  transpose_to_bf16_kernel<<<256, 256, 0, stream>>>(pe_w2,  pe_w2T,  256, 256);
  transpose_to_bf16_kernel<<<256, 256, 0, stream>>>(ps_w2,  ps_w2T,  256, 256);
  transpose_to_bf16_kernel<<<256, 256, 0, stream>>>(proj_p, proj_pT, 256, 256);
  transpose_to_bf16_kernel<<<256, 256, 0, stream>>>(sig_w1, sig_w1T, 256, 256);

  hipMemsetAsync(lg, 0, 512 * 4, stream);

  embed_landmarks_kernel<<<N_L / 16, 256, 0, stream>>>(
      l_x, l_typ, le_emb, le_w1, le_b1, le_b2, le_g, le_be, le_w2T, proj_lT, hL, hLb);
  embed_points_kernel<<<N_P / 16, 256, 0, stream>>>(
      p_x, p_score, p_rad, p_typ, pe_emb, pe_b1, pe_b2, ps_w1, ps_b1, ps_b2,
      pe_g, pe_be, sig_b1, sig_w2, sig_b2,
      pe_w1T, pe_w2T, ps_w2T, proj_pT, sig_w1T, hP, hPb, p_sigma);
  logits_kernel<<<dim3(N_P / 128, N_L / 32), 256, 0, stream>>>(hLb, hPb, logits);
  row_stats_kernel<<<N_L, 256, 0, stream>>>(logits, hL, dustbin, sharp_r, ent_r,
                                            topj, logWtop);
  colsum_kernel<<<N_L / 16 + N_P / 16, 256, 0, stream>>>(hL, hP, lg, pg);
  finalize_kernel<<<1, 256, 0, stream>>>(
      sharp_r, ent_r, lg, pg, retr_l, retr_p, l_pos, p_pos, pairs,
      topj, logWtop, p_sigma, hi_w1, hi_b1, hi_w2, hi_b2, (float*)d_out);
}
